// Conv2d_int8_T_50483045597925
// MI455X (gfx1250) — 8-bit matrix operand order repaired and checked by reading, compiled, not run on hardware
//
#include <hip/hip_runtime.h>

typedef __attribute__((ext_vector_type(8))) int v8i;

#define CIN    64
#define COUT   64
#define HH     28
#define WW     28
#define BATCH  8
#define KDIM   576
#define KT     9
#define NPIX   784
#define NTOT   6272
#define NT     392
#define MT     4
#define NX     401408
#define NW     36864

__device__ __forceinline__ float scale_x(const unsigned* mb) {
    float mx = __uint_as_float(mb[0]);
    return (2.85f + 0.05f * mx) / 127.0f;
}
__device__ __forceinline__ float scale_w(const unsigned* mb) {
    float mw = __uint_as_float(mb[1]);
    return (0.285f + 0.05f * mw) / 127.0f;
}
__device__ __forceinline__ int quant8(float v, float s) {
    float r = rintf(v / s);
    r = fminf(127.0f, fmaxf(-128.0f, r));
    return (int)r;
}

__global__ void k_init(unsigned* mb) {
    if (threadIdx.x < 2) mb[threadIdx.x] = 0u;
}

__global__ void __launch_bounds__(256) k_absmax(const float* __restrict__ x,
                                                const float* __restrict__ w,
                                                unsigned* mb) {
    __shared__ float smx[256];
    __shared__ float smw[256];
    float mx = 0.0f, mw = 0.0f;
    int stride = gridDim.x * blockDim.x;
    for (int i = blockIdx.x * blockDim.x + threadIdx.x; i < NX; i += stride)
        mx = fmaxf(mx, fabsf(x[i]));
    for (int i = blockIdx.x * blockDim.x + threadIdx.x; i < NW; i += stride)
        mw = fmaxf(mw, fabsf(w[i]));
    smx[threadIdx.x] = mx;
    smw[threadIdx.x] = mw;
    __syncthreads();
    for (int s = 128; s > 0; s >>= 1) {
        if (threadIdx.x < s) {
            smx[threadIdx.x] = fmaxf(smx[threadIdx.x], smx[threadIdx.x + s]);
            smw[threadIdx.x] = fmaxf(smw[threadIdx.x], smw[threadIdx.x + s]);
        }
        __syncthreads();
    }
    if (threadIdx.x == 0) {
        atomicMax(&mb[0], __float_as_uint(smx[0]));
        atomicMax(&mb[1], __float_as_uint(smw[0]));
    }
}

__global__ void __launch_bounds__(256) k_packA(const float* __restrict__ w,
                                               const unsigned* __restrict__ mb,
                                               int* __restrict__ packA) {
    int idx = blockIdx.x * blockDim.x + threadIdx.x;
    float sw = scale_w(mb);
    int v    = idx & 7;
    int lane = (idx >> 3) & 31;
    int t    = idx >> 8;
    int kt   = t % 9;
    int mt   = t / 9;
    int m    = mt * 16 + (lane & 15);
    int h    = lane >> 4;
    int kb   = kt * 64 + (v >> 1) * 16 + h * 8 + (v & 1) * 4;
    unsigned pk = 0u;
#pragma unroll
    for (int j = 0; j < 4; ++j) {
        int q = quant8(w[m * KDIM + kb + j], sw);
        pk |= ((unsigned)(q & 255)) << (8 * j);
    }
    packA[idx] = (int)pk;
}

__global__ void __launch_bounds__(256) k_packB(const float* __restrict__ x,
                                               const unsigned* __restrict__ mb,
                                               int* __restrict__ packB) {
    int idx = blockIdx.x * blockDim.x + threadIdx.x;
    float sx = scale_x(mb);
    int v    = idx & 7;
    int lane = (idx >> 3) & 31;
    int t    = idx >> 8;
    int kt   = t % 9;
    int nt   = t / 9;
    int n    = nt * 16 + (lane & 15);
    int h    = lane >> 4;
    int kb   = kt * 64 + (v >> 1) * 16 + h * 8 + (v & 1) * 4;
    int b    = n / NPIX;
    int p    = n - b * NPIX;
    int oh   = p / WW;
    int ow   = p - oh * WW;
    unsigned pk = 0u;
#pragma unroll
    for (int j = 0; j < 4; ++j) {
        int k  = kb + j;
        int ci = k / 9;
        int r  = k - ci * 9;
        int kh = r / 3;
        int kw = r - kh * 3;
        int ih = oh + kh - 1;
        int iw = ow + kw - 1;
        int q = 0;
        if (ih >= 0 && ih < HH && iw >= 0 && iw < WW)
            q = quant8(x[((b * CIN + ci) * HH + ih) * WW + iw], sx);
        pk |= ((unsigned)(q & 255)) << (8 * j);
    }
    packB[idx] = (int)pk;
}

__global__ void __launch_bounds__(256) k_gemm(const int* __restrict__ packA,
                                              const int* __restrict__ packB,
                                              const float* __restrict__ bias,
                                              const unsigned* __restrict__ mb,
                                              float* __restrict__ out) {
    int lane = threadIdx.x & 31;
    int tile = blockIdx.x * 8 + (threadIdx.x >> 5);
    int mt   = tile & 3;
    int nt   = tile >> 2;

    const int* ap = packA + (mt * 9) * 256 + lane * 8;
    const int* bp = packB + (nt * 9) * 256 + lane * 8;

    v8i c = {};
#pragma unroll
    for (int kt = 0; kt < 9; ++kt) {
        v8i a = *(const v8i*)(ap + kt * 256);
        v8i b = *(const v8i*)(bp + kt * 256);
        c = __builtin_amdgcn_wmma_i32_16x16x64_iu8(true, a, true, b, c, false, false);
    }

    float s = scale_x(mb) * scale_w(mb);
    int n    = nt * 16 + (lane & 15);
    int bimg = n / NPIX;
    int p    = n - bimg * NPIX;
    int mbase = mt * 16 + (lane >> 4) * 8;
#pragma unroll
    for (int d = 0; d < 8; ++d) {
        int m = mbase + d;
        out[(bimg * COUT + m) * NPIX + p] = (float)c[d] * s + bias[m];
    }
}

extern "C" void kernel_launch(void* const* d_in, const int* in_sizes, int n_in,
                              void* d_out, int out_size, void* d_ws, size_t ws_size,
                              hipStream_t stream) {
    const float* x    = (const float*)d_in[0];
    const float* w    = (const float*)d_in[1];
    const float* bias = (const float*)d_in[2];

    unsigned* mb  = (unsigned*)d_ws;
    int* packA    = (int*)((char*)d_ws + 256);
    int* packB    = (int*)((char*)d_ws + 40960);
    float* out    = (float*)d_out;

    k_init  <<<1,    32, 0, stream>>>(mb);
    k_absmax<<<128, 256, 0, stream>>>(x, w, mb);
    k_packA <<<36,  256, 0, stream>>>(w, mb, packA);
    k_packB <<<3528,256, 0, stream>>>(x, mb, packB);
    k_gemm  <<<196, 256, 0, stream>>>(packA, packB, bias, mb, out);
}
